// SimplePICOContrastiveRAG_37538014167093
// MI455X (gfx1250) — matrix operand order repaired and checked by reading, not compiled, not run on hardware
//
#include <hip/hip_runtime.h>
#include <math.h>
#include <stdint.h>

// ---------------------------------------------------------------------------
// Types for CDNA5 WMMA (wave32, 16x16x32 bf16 -> f32)
// ---------------------------------------------------------------------------
typedef __bf16 bf16_t;
typedef __attribute__((ext_vector_type(16))) __bf16 v16bf;
typedef __attribute__((ext_vector_type(8)))  __bf16 v8bf;
typedef __attribute__((ext_vector_type(8)))  float  v8f;

__device__ __forceinline__ v8f wmma_bf16(v16bf a, v16bf b, v8f c) {
  // 8 args: (neg_a, A, neg_b, B, c_mod, C, reuse_a, reuse_b)
  return __builtin_amdgcn_wmma_f32_16x16x32_bf16(false, a, false, b, (short)0, c,
                                                 false, false);
}

__device__ __forceinline__ v16bf cat8(v8bf lo, v8bf hi) {
  return __builtin_shufflevector(lo, hi, 0, 1, 2, 3, 4, 5, 6, 7,
                                 8, 9, 10, 11, 12, 13, 14, 15);
}

// CDNA5 async Global->LDS copy (ASYNCcnt-tracked, CDNA5 ISA).
// lds_off: byte offset in LDS (low 32 bits of the flat LDS aperture address).
__device__ __forceinline__ void async_ld_b128(unsigned lds_off,
                                              const void* gaddr) {
  asm volatile("global_load_async_to_lds_b128 %0, %1, off"
               :
               : "v"(lds_off), "v"(gaddr)
               : "memory");
}

template <int N>
__device__ __forceinline__ void wait_asynccnt() {
  asm volatile("s_wait_asynccnt %0" ::"i"(N) : "memory");
}

// ---------------------------------------------------------------------------
// Problem constants
// ---------------------------------------------------------------------------
#define BSZ   2048
#define NCORP 100000
#define EDIM  512
#define HDIM  1024
#define CDIM  256
#define TDIM  64
#define KTOP  16

#define CHUNK   1024
#define NCHUNK  98                 // ceil(100000 / 1024)
#define NPAD    (NCHUNK * CHUNK)   // 100352, zero-padded corpus rows
#define CAND    (NCHUNK * KTOP)    // 1568 merge candidates per row

#define NEG_INF (-3.0e38f)

// d_out element offsets (float units): outcome | scores | indices | contrastive
#define OUT_OUTCOME 0
#define OUT_SCORES  (BSZ)                       // 2048
#define OUT_IDX     (OUT_SCORES + BSZ * KTOP)   // 34816
#define OUT_CEMB    (OUT_IDX + BSZ * KTOP)      // 67584

// GEMM epilogue flags
#define GF_RELU 1
#define GF_F32  2
#define GF_OUT2 4

// ---------------------------------------------------------------------------
// Elementwise f32 -> bf16
// ---------------------------------------------------------------------------
__global__ void f2bf_kernel(const float* __restrict__ in, bf16_t* __restrict__ out,
                            int n) {
  int i = blockIdx.x * 256 + threadIdx.x;
  if (i < n) out[i] = (bf16_t)in[i];
}

// Transpose-convert weight W[K,N] (row-major f32) -> Wt[N,K] (row-major bf16)
__global__ void t2bf_kernel(const float* __restrict__ in, bf16_t* __restrict__ out,
                            int K, int N) {
  int i = blockIdx.x * 256 + threadIdx.x;
  if (i < K * N) {
    int n = i / K;
    int k = i - n * K;
    out[i] = (bf16_t)in[(size_t)k * N + n];
  }
}

// Corpus: bf16 copy (zero-padded to NPAD rows) + per-row 1/max(||c||,eps)
__global__ void corpus_prep_kernel(const float* __restrict__ corpus,
                                   bf16_t* __restrict__ cb,
                                   float* __restrict__ invn) {
  int wave = threadIdx.x >> 5, lane = threadIdx.x & 31;
  int row = blockIdx.x * 4 + wave;
  if (row >= NPAD) return;
  bf16_t* o = cb + (size_t)row * EDIM;
  if (row < NCORP) {
    const float* x = corpus + (size_t)row * EDIM;
    float s = 0.f;
    for (int e = lane; e < EDIM; e += 32) {
      float v = x[e];
      s += v * v;
      o[e] = (bf16_t)v;
    }
    for (int off = 16; off > 0; off >>= 1) s += __shfl_down(s, off);
    if (lane == 0) invn[row] = 1.0f / fmaxf(sqrtf(s), 1e-12f);
  } else {
    for (int e = lane; e < EDIM; e += 32) o[e] = (bf16_t)0.f;
    if (lane == 0) invn[row] = 0.f;
  }
}

// L2-normalize contrastive embedding (f32, already in d_out) -> bf16 query
__global__ void norm_query_kernel(const float* __restrict__ cemb,
                                  bf16_t* __restrict__ q) {
  int wave = threadIdx.x >> 5, lane = threadIdx.x & 31;
  int row = blockIdx.x * 4 + wave;
  const float* x = cemb + (size_t)row * EDIM;
  float s = 0.f;
  for (int e = lane; e < EDIM; e += 32) {
    float v = x[e];
    s += v * v;
  }
  for (int off = 16; off > 0; off >>= 1) s += __shfl_down(s, off);
  s = __shfl(s, 0);
  float inv = 1.0f / fmaxf(sqrtf(s), 1e-12f);
  for (int e = lane; e < EDIM; e += 32)
    q[(size_t)row * EDIM + e] = (bf16_t)(x[e] * inv);
}

// ---------------------------------------------------------------------------
// Generic bf16 WMMA GEMM: out[M,N] = act(A[M,K] * Bt[N,K]^T + bias)
// Block = 128 threads (4 waves) -> 64x64 output tile; each wave 16x64.
// The 64-col B k-panel (4 KB / 32-k chunk) is shared by all 4 waves, so it is
// double-buffered through LDS with CDNA5 async global->LDS loads (ASYNCcnt).
// A rows are per-wave unique and stay on the direct global_load_b128 path.
// ---------------------------------------------------------------------------
__device__ __forceinline__ void store_tile(v8f acc, int row0, int c,
                                           const float* __restrict__ bias,
                                           void* __restrict__ out, int ldo,
                                           int col_off,
                                           void* __restrict__ out2, int ldo2,
                                           int col_off2, int flags) {
  float bv = bias ? bias[c] : 0.f;
#pragma unroll
  for (int v = 0; v < 8; ++v) {
    int row = row0 + v;
    float val = acc[v] + bv;
    if (flags & GF_RELU) val = fmaxf(val, 0.f);
    if (flags & GF_F32) {
      ((float*)out)[(size_t)row * ldo + col_off + c] = val;
    } else {
      bf16_t o = (bf16_t)val;
      ((bf16_t*)out)[(size_t)row * ldo + col_off + c] = o;
      if (flags & GF_OUT2)
        ((bf16_t*)out2)[(size_t)row * ldo2 + col_off2 + c] = o;
    }
  }
}

__global__ __launch_bounds__(128) void gemm_bf16_kernel(
    const bf16_t* __restrict__ A, int lda, const bf16_t* __restrict__ Bt,
    int ldb, const float* __restrict__ bias, void* __restrict__ out, int ldo,
    int col_off, void* __restrict__ out2, int ldo2, int col_off2, int K,
    int flags) {
  // [buf][col*32 + k] : per column 32 contiguous bf16 (one 32-wide k chunk)
  __shared__ __align__(64) bf16_t sB[2][64 * 32];  // 2 x 4 KB

  int tid = threadIdx.x;
  int lane = tid & 31;
  int wave = tid >> 5;
  int l15 = lane & 15;
  int kHalf = lane >> 4;  // A frag: 0 -> K 0..15, 1 -> K 16..31
  int rowBase = (blockIdx.y * 4 + wave) * 16;
  int colBase = blockIdx.x * 64;

  // Async-copy slice for this thread: 16 bf16 (32 B) of the 4 KB panel.
  int pcol = tid >> 1;   // 0..63
  int phalf = tid & 1;   // which 16-element half of the column's 32-k chunk
  const bf16_t* gsrc =
      Bt + (size_t)(colBase + pcol) * ldb + phalf * 16;
  unsigned ldsoff0 = (unsigned)(uintptr_t)&sB[0][pcol * 32 + phalf * 16];
  unsigned ldsoff1 = (unsigned)(uintptr_t)&sB[1][pcol * 32 + phalf * 16];

  // Prologue: start chunk 0 into buffer 0.
  async_ld_b128(ldsoff0, gsrc);
  async_ld_b128(ldsoff0 + 16, gsrc + 8);

  const bf16_t* arow = A + (size_t)(rowBase + l15) * lda + kHalf * 16;
  v8f acc0 = {}, acc1 = {}, acc2 = {}, acc3 = {};

  int nk = K >> 5;
  for (int i = 0; i < nk; ++i) {
    __syncthreads();  // all waves done reading buffer (i+1)&1 -> overwritable
    if (i + 1 < nk) {
      const bf16_t* g = gsrc + (size_t)(i + 1) * 32;
      unsigned lo = ((i + 1) & 1) ? ldsoff1 : ldsoff0;
      async_ld_b128(lo, g);
      async_ld_b128(lo + 16, g + 8);
      wait_asynccnt<2>();  // chunk i complete (in-order); chunk i+1 in flight
    } else {
      wait_asynccnt<0>();
    }
    __syncthreads();  // chunk i visible to all waves

    int k0 = i << 5;
    v16bf a = cat8(*(const v8bf*)(arow + k0), *(const v8bf*)(arow + k0 + 8));
    const bf16_t* sb = &sB[i & 1][kHalf * 16];
    v16bf b0 = *(const v16bf*)(sb + (l15 + 0) * 32);
    v16bf b1 = *(const v16bf*)(sb + (l15 + 16) * 32);
    v16bf b2 = *(const v16bf*)(sb + (l15 + 32) * 32);
    v16bf b3 = *(const v16bf*)(sb + (l15 + 48) * 32);
    acc0 = wmma_bf16(a, b0, acc0);
    acc1 = wmma_bf16(a, b1, acc1);
    acc2 = wmma_bf16(a, b2, acc2);
    acc3 = wmma_bf16(a, b3, acc3);
  }

  int row0 = rowBase + kHalf * 8;
  store_tile(acc0, row0, colBase + 0 + l15, bias, out, ldo, col_off, out2, ldo2,
             col_off2, flags);
  store_tile(acc1, row0, colBase + 16 + l15, bias, out, ldo, col_off, out2,
             ldo2, col_off2, flags);
  store_tile(acc2, row0, colBase + 32 + l15, bias, out, ldo, col_off, out2,
             ldo2, col_off2, flags);
  store_tile(acc3, row0, colBase + 48 + l15, bias, out, ldo, col_off, out2,
             ldo2, col_off2, flags);
}

// ---------------------------------------------------------------------------
// Fused similarity GEMM + per-chunk top-16.
// grid = (NCHUNK, BSZ/16); block = 128 (4 waves).
// Each block: 16 query rows x 1024 corpus cols. Scores staged in 64KB LDS,
// then threads 0..15 maintain a sorted top-16 per row.
// Corpus stays bf16/unnormalized; scale by invn[c] post-accumulation.
// Corpus (102 MB bf16) is L2-resident on MI455X (192 MB global L2), so the
// 128 row-tile sweeps hit L2 after the first pass.
// ---------------------------------------------------------------------------
__global__ __launch_bounds__(128) void sim_topk_kernel(
    const bf16_t* __restrict__ Q, const bf16_t* __restrict__ C,
    const float* __restrict__ invn, float* __restrict__ pvals,
    int* __restrict__ pidx) {
  __shared__ float s[16][CHUNK];  // 64 KB

  int tid = threadIdx.x;
  int lane = tid & 31, wave = tid >> 5;
  int l15 = lane & 15, kHalf = lane >> 4;
  int rowBase = blockIdx.y * 16;
  int chunkBase = blockIdx.x * CHUNK;

  // Preload all 16 A fragments for this 16-row query tile (E=512 -> 16 steps).
  v16bf a[16];
  const bf16_t* qrow = Q + (size_t)(rowBase + l15) * EDIM + kHalf * 16;
#pragma unroll
  for (int kk = 0; kk < 16; ++kk) {
    const bf16_t* ap = qrow + kk * 32;
    a[kk] = cat8(*(const v8bf*)ap, *(const v8bf*)(ap + 8));
  }

  for (int t = wave; t < CHUNK / 16; t += 4) {
    int c = chunkBase + t * 16 + l15;  // B-fragment column for this lane
    const bf16_t* brow = C + (size_t)c * EDIM + kHalf * 16;
    v8f acc = {};
#pragma unroll
    for (int kk = 0; kk < 16; ++kk) {
      v16bf b = *(const v16bf*)(brow + kk * 32);
      acc = wmma_bf16(a[kk], b, acc);
    }
    float sc = (c < NCORP) ? invn[c] : 0.f;
    bool valid = (c < NCORP);
#pragma unroll
    for (int v = 0; v < 8; ++v) {
      float val = valid ? acc[v] * sc : NEG_INF;
      s[v + kHalf * 8][t * 16 + l15] = val;
    }
  }
  __syncthreads();

  if (tid < 16) {
    float vals[KTOP];
    int idxs[KTOP];
#pragma unroll
    for (int r = 0; r < KTOP; ++r) {
      vals[r] = NEG_INF;
      idxs[r] = 0;
    }
    for (int j = 0; j < CHUNK; ++j) {
      float v = s[tid][j];
      if (v > vals[KTOP - 1]) {
        int p = KTOP - 1;
        while (p > 0 && vals[p - 1] < v) {
          vals[p] = vals[p - 1];
          idxs[p] = idxs[p - 1];
          --p;
        }
        vals[p] = v;
        idxs[p] = chunkBase + j;
      }
    }
    size_t base = ((size_t)(rowBase + tid) * NCHUNK + blockIdx.x) * KTOP;
    for (int r = 0; r < KTOP; ++r) {
      pvals[base + r] = vals[r];
      pidx[base + r] = idxs[r];
    }
  }
}

// Merge 1568 candidates/row -> sorted top-16. One wave per row.
__global__ void merge_topk_kernel(const float* __restrict__ pvals,
                                  const int* __restrict__ pidx,
                                  float* __restrict__ oscore,
                                  int* __restrict__ oidx) {
  __shared__ float sv[CAND];
  __shared__ int si[CAND];
  int row = blockIdx.x;
  int lane = threadIdx.x;
  size_t base = (size_t)row * CAND;
  for (int j = lane; j < CAND; j += 32) {
    sv[j] = pvals[base + j];
    si[j] = pidx[base + j];
  }
  __syncthreads();
  for (int r = 0; r < KTOP; ++r) {
    float best = NEG_INF;
    int bpos = 0;
    for (int j = lane; j < CAND; j += 32) {
      float v = sv[j];
      if (v > best) {
        best = v;
        bpos = j;
      }
    }
    for (int off = 16; off > 0; off >>= 1) {
      float ov = __shfl_down(best, off);
      int op = __shfl_down(bpos, off);
      if (ov > best) {
        best = ov;
        bpos = op;
      }
    }
    if (lane == 0) {
      oscore[row * KTOP + r] = best;
      oidx[row * KTOP + r] = si[bpos];
      sv[bpos] = NEG_INF;
    }
    __syncthreads();
  }
}

// Gather retrieved corpus rows (un-normalized, bf16) into flat [B, K*E].
// One thread copies 8 bf16 (16 bytes). Exactly 2048*16*64 threads.
__global__ void gather_flat_kernel(const bf16_t* __restrict__ cb,
                                   const int* __restrict__ idx,
                                   bf16_t* __restrict__ flat) {
  int i = blockIdx.x * 256 + threadIdx.x;
  int e8 = i & 63;
  int kk = (i >> 6) & 15;
  int b = i >> 10;
  int r = idx[b * KTOP + kk];
  const v8bf* src = (const v8bf*)(cb + (size_t)r * EDIM + e8 * 8);
  *(v8bf*)(flat + (size_t)b * (KTOP * EDIM) + (size_t)kk * EDIM + e8 * 8) = *src;
}

// outcome[row] = z2[row,:512] . w3[:512,0] + b3 ; one wave per row
__global__ void final_dot_kernel(const bf16_t* __restrict__ z2,
                                 const float* __restrict__ w3,
                                 const float* __restrict__ b3,
                                 float* __restrict__ out) {
  int wave = threadIdx.x >> 5, lane = threadIdx.x & 31;
  int row = blockIdx.x * 4 + wave;
  const bf16_t* zr = z2 + (size_t)row * 512;
  float s = 0.f;
  for (int e = lane; e < 512; e += 32) s += (float)zr[e] * w3[e];
  for (int off = 16; off > 0; off >>= 1) s += __shfl_down(s, off);
  if (lane == 0) out[row] = s + b3[0];
}

// ---------------------------------------------------------------------------
// Host launcher
// ---------------------------------------------------------------------------
extern "C" void kernel_launch(void* const* d_in, const int* in_sizes, int n_in,
                              void* d_out, int out_size, void* d_ws,
                              size_t ws_size, hipStream_t stream) {
  const float* patient = (const float*)d_in[0];    // [B, 320]
  const float* treatment = (const float*)d_in[1];  // [B, 64]
  const float* confound = (const float*)d_in[2];   // [B, 256]
  const float* corpus = (const float*)d_in[3];     // [N, 512]
  const float* pe_w1 = (const float*)d_in[4];      // [320,1024]
  const float* pe_b1 = (const float*)d_in[5];
  const float* pe_w2 = (const float*)d_in[6];      // [1024,1024]
  const float* pe_b2 = (const float*)d_in[7];
  const float* te_w = (const float*)d_in[8];       // [64,1024]
  const float* te_b = (const float*)d_in[9];
  const float* ce_w = (const float*)d_in[10];      // [256,1024]
  const float* ce_b = (const float*)d_in[11];
  const float* co_w1 = (const float*)d_in[12];     // [2048,1024]
  const float* co_b1 = (const float*)d_in[13];
  const float* co_w2 = (const float*)d_in[14];     // [1024,512]
  const float* co_b2 = (const float*)d_in[15];
  const float* re_w = (const float*)d_in[16];      // [8192,1024]
  const float* re_b = (const float*)d_in[17];
  const float* op_w1 = (const float*)d_in[18];     // [3072,1024]
  const float* op_b1 = (const float*)d_in[19];
  const float* op_w2 = (const float*)d_in[20];     // [1024,512]
  const float* op_b2 = (const float*)d_in[21];
  const float* op_w3 = (const float*)d_in[22];     // [512,1]
  const float* op_b3 = (const float*)d_in[23];

  float* out_f = (float*)d_out;
  float* out_outcome = out_f + OUT_OUTCOME;
  float* out_scores = out_f + OUT_SCORES;
  int* out_idx = (int*)(out_f + OUT_IDX);
  float* out_cemb = out_f + OUT_CEMB;

  // Workspace carve-up
  char* ws = (char*)d_ws;
  size_t off = 0;
  auto alloc = [&](size_t bytes) -> void* {
    void* p = ws + off;
    off = (off + bytes + 255) & ~(size_t)255;
    return p;
  };
  bf16_t* corpus_bf = (bf16_t*)alloc((size_t)NPAD * EDIM * 2);  // 102.8 MB
  float* invn = (float*)alloc((size_t)NPAD * 4);
  bf16_t* patient_bf = (bf16_t*)alloc((size_t)BSZ * 320 * 2);
  bf16_t* treat_bf = (bf16_t*)alloc((size_t)BSZ * 64 * 2);
  bf16_t* conf_bf = (bf16_t*)alloc((size_t)BSZ * 256 * 2);
  bf16_t* pe_w1t = (bf16_t*)alloc((size_t)1024 * 320 * 2);
  bf16_t* pe_w2t = (bf16_t*)alloc((size_t)1024 * 1024 * 2);
  bf16_t* te_wt = (bf16_t*)alloc((size_t)1024 * 64 * 2);
  bf16_t* ce_wt = (bf16_t*)alloc((size_t)1024 * 256 * 2);
  bf16_t* co_w1t = (bf16_t*)alloc((size_t)1024 * 2048 * 2);
  bf16_t* co_w2t = (bf16_t*)alloc((size_t)512 * 1024 * 2);
  bf16_t* re_wt = (bf16_t*)alloc((size_t)1024 * 8192 * 2);
  bf16_t* op_w1t = (bf16_t*)alloc((size_t)1024 * 3072 * 2);
  bf16_t* op_w2t = (bf16_t*)alloc((size_t)512 * 1024 * 2);
  bf16_t* h1 = (bf16_t*)alloc((size_t)BSZ * 1024 * 2);
  bf16_t* hcat = (bf16_t*)alloc((size_t)BSZ * 2048 * 2);     // [pat|conf]
  bf16_t* co_h = (bf16_t*)alloc((size_t)BSZ * 1024 * 2);
  bf16_t* query_bf = (bf16_t*)alloc((size_t)BSZ * 512 * 2);
  bf16_t* combined = (bf16_t*)alloc((size_t)BSZ * 3072 * 2); // [tre|conf|ret]
  bf16_t* flat = (bf16_t*)alloc((size_t)BSZ * 8192 * 2);
  bf16_t* z1 = (bf16_t*)alloc((size_t)BSZ * 1024 * 2);
  bf16_t* z2 = (bf16_t*)alloc((size_t)BSZ * 512 * 2);
  float* pvals = (float*)alloc((size_t)BSZ * CAND * 4);
  int* pidx = (int*)alloc((size_t)BSZ * CAND * 4);
  (void)ws_size;

  auto cdiv = [](int a, int b) { return (a + b - 1) / b; };

  // --- Precision conversions (inputs + transposed weights) ---
  f2bf_kernel<<<cdiv(BSZ * 320, 256), 256, 0, stream>>>(patient, patient_bf, BSZ * 320);
  f2bf_kernel<<<cdiv(BSZ * 64, 256), 256, 0, stream>>>(treatment, treat_bf, BSZ * 64);
  f2bf_kernel<<<cdiv(BSZ * 256, 256), 256, 0, stream>>>(confound, conf_bf, BSZ * 256);
  t2bf_kernel<<<cdiv(320 * 1024, 256), 256, 0, stream>>>(pe_w1, pe_w1t, 320, 1024);
  t2bf_kernel<<<cdiv(1024 * 1024, 256), 256, 0, stream>>>(pe_w2, pe_w2t, 1024, 1024);
  t2bf_kernel<<<cdiv(64 * 1024, 256), 256, 0, stream>>>(te_w, te_wt, 64, 1024);
  t2bf_kernel<<<cdiv(256 * 1024, 256), 256, 0, stream>>>(ce_w, ce_wt, 256, 1024);
  t2bf_kernel<<<cdiv(2048 * 1024, 256), 256, 0, stream>>>(co_w1, co_w1t, 2048, 1024);
  t2bf_kernel<<<cdiv(1024 * 512, 256), 256, 0, stream>>>(co_w2, co_w2t, 1024, 512);
  t2bf_kernel<<<cdiv(8192 * 1024, 256), 256, 0, stream>>>(re_w, re_wt, 8192, 1024);
  t2bf_kernel<<<cdiv(3072 * 1024, 256), 256, 0, stream>>>(op_w1, op_w1t, 3072, 1024);
  t2bf_kernel<<<cdiv(1024 * 512, 256), 256, 0, stream>>>(op_w2, op_w2t, 1024, 512);
  corpus_prep_kernel<<<NPAD / 4, 128, 0, stream>>>(corpus, corpus_bf, invn);

  // --- Encoder GEMMs (all bf16 WMMA, async-LDS B staging) ---
  // h1 = relu(patient @ pe_w1 + b1)
  gemm_bf16_kernel<<<dim3(1024 / 64, BSZ / 64), 128, 0, stream>>>(
      patient_bf, 320, pe_w1t, 320, pe_b1, h1, 1024, 0, nullptr, 0, 0, 320, GF_RELU);
  // patient_emb -> hcat[:, 0:1024]
  gemm_bf16_kernel<<<dim3(1024 / 64, BSZ / 64), 128, 0, stream>>>(
      h1, 1024, pe_w2t, 1024, pe_b2, hcat, 2048, 0, nullptr, 0, 0, 1024, 0);
  // confounder_emb -> hcat[:, 1024:2048] AND combined[:, 1024:2048]
  gemm_bf16_kernel<<<dim3(1024 / 64, BSZ / 64), 128, 0, stream>>>(
      conf_bf, 256, ce_wt, 256, ce_b, hcat, 2048, 1024, combined, 3072, 1024,
      256, GF_OUT2);
  // treatment_emb -> combined[:, 0:1024]
  gemm_bf16_kernel<<<dim3(1024 / 64, BSZ / 64), 128, 0, stream>>>(
      treat_bf, 64, te_wt, 64, te_b, combined, 3072, 0, nullptr, 0, 0, 64, 0);
  // co_h = relu(hcat @ co_w1 + b)
  gemm_bf16_kernel<<<dim3(1024 / 64, BSZ / 64), 128, 0, stream>>>(
      hcat, 2048, co_w1t, 2048, co_b1, co_h, 1024, 0, nullptr, 0, 0, 2048, GF_RELU);
  // contrastive_emb (f32, straight into d_out)
  gemm_bf16_kernel<<<dim3(512 / 64, BSZ / 64), 128, 0, stream>>>(
      co_h, 1024, co_w2t, 1024, co_b2, out_cemb, 512, 0, nullptr, 0, 0, 1024, GF_F32);

  // --- Retrieval: normalize, fused similarity+topk, merge, gather ---
  norm_query_kernel<<<BSZ / 4, 128, 0, stream>>>(out_cemb, query_bf);
  sim_topk_kernel<<<dim3(NCHUNK, BSZ / 16), 128, 0, stream>>>(
      query_bf, corpus_bf, invn, pvals, pidx);
  merge_topk_kernel<<<BSZ, 32, 0, stream>>>(pvals, pidx, out_scores, out_idx);
  gather_flat_kernel<<<(BSZ * KTOP * 64) / 256, 256, 0, stream>>>(
      corpus_bf, out_idx, flat);

  // retrieval_enc -> combined[:, 2048:3072]
  gemm_bf16_kernel<<<dim3(1024 / 64, BSZ / 64), 128, 0, stream>>>(
      flat, 8192, re_wt, 8192, re_b, combined, 3072, 2048, nullptr, 0, 0, 8192, 0);

  // --- Outcome head ---
  gemm_bf16_kernel<<<dim3(1024 / 64, BSZ / 64), 128, 0, stream>>>(
      combined, 3072, op_w1t, 3072, op_b1, z1, 1024, 0, nullptr, 0, 0, 3072, GF_RELU);
  gemm_bf16_kernel<<<dim3(512 / 64, BSZ / 64), 128, 0, stream>>>(
      z1, 1024, op_w2t, 1024, op_b2, z2, 512, 0, nullptr, 0, 0, 1024, GF_RELU);
  final_dot_kernel<<<BSZ / 4, 128, 0, stream>>>(z2, op_w3, op_b3, out_outcome);
}
